// FirstLayer_22050362098048
// MI455X (gfx1250) — matrix operand order repaired and checked by reading, not compiled, not run on hardware
//
#include <hip/hip_runtime.h>
#include <hip/hip_bf16.h>

// ---------------------------------------------------------------------------
// LayerNorm-LSTM for MI455X (gfx1250, wave32, WMMA, TDM).
// B=64, S=512, D=1024.  comb = LN([x_t, h] @ W + b); gates; c/h recurrence.
// Persistent kernel (64 WGs x 128 thr). Each WG owns 16 output units ->
// 64 W-columns x 2048 K in bf16 = 256 KB, resident in LDS (320 KB/WGP),
// loaded once via tensor_load_to_lds (TDM) and consumed by
// v_wmma_f32_16x16x32_bf16 every step. Two grid barriers per step.
// ---------------------------------------------------------------------------

typedef __attribute__((ext_vector_type(16))) __bf16 v16bf;
typedef __attribute__((ext_vector_type(8)))  __bf16 v8bf;
typedef __attribute__((ext_vector_type(8)))  float  v8f;
typedef __attribute__((ext_vector_type(4)))  unsigned int v4u;
typedef __attribute__((ext_vector_type(8)))  int v8i;
typedef __attribute__((ext_vector_type(4)))  int v4i;

#define B_   64
#define S_   512
#define D_   1024
#define N4   4096      // 4*D columns of W
#define K2   2048      // 2*D reduction dim
#define NWG  64        // persistent workgroups
#define JPW  16        // output units per WG (one 16-wide j-tile)
#define NTHR 128       // 4 wave32 = 4 M-tiles of 16 batch rows

union AFrag { v16bf v; struct { v8bf lo, hi; } p; };

// ---------------------------------------------------------------------------
// prep kernels
// ---------------------------------------------------------------------------
__global__ void cvt_x_kernel(const float* __restrict__ X,
                             __bf16* __restrict__ Xb, int n) {
    int i = blockIdx.x * blockDim.x + threadIdx.x;
    int stride = gridDim.x * blockDim.x;
    for (; i < n; i += stride) Xb[i] = (__bf16)X[i];
}

// W: [K2][N4] row-major  ->  Wt: [N4][K2] bf16 (column-major of W)
__global__ void cvt_w_kernel(const float* __restrict__ W,
                             __bf16* __restrict__ Wt) {
    int i = blockIdx.x * blockDim.x + threadIdx.x;
    int stride = gridDim.x * blockDim.x;
    const int n = N4 * K2;
    for (; i < n; i += stride) {
        int col = i / K2;
        int k   = i - col * K2;
        Wt[i] = (__bf16)W[(size_t)k * N4 + col];
    }
}

__global__ void init_state_kernel(const float* __restrict__ hx,
                                  const float* __restrict__ cx,
                                  __bf16* __restrict__ Hb,
                                  float* __restrict__ Cs,
                                  float* __restrict__ part,
                                  unsigned* __restrict__ ctr) {
    int i = blockIdx.x * blockDim.x + threadIdx.x;
    int stride = gridDim.x * blockDim.x;
    for (int idx = i; idx < B_ * D_; idx += stride) {
        int j = idx & (D_ - 1);
        Hb[idx] = (__bf16)hx[j];
        Cs[idx] = cx[j];
    }
    for (int idx = i; idx < NWG * 2 * B_; idx += stride) part[idx] = 0.0f;
    if (i == 0) *ctr = 0u;
}

// ---------------------------------------------------------------------------
// grid-wide barrier (monotonic counter, reset each launch by init kernel)
// ---------------------------------------------------------------------------
__device__ __forceinline__ void grid_sync(unsigned* ctr, unsigned target) {
    __threadfence();
    __syncthreads();
    if (threadIdx.x == 0) {
        __atomic_fetch_add(ctr, 1u, __ATOMIC_ACQ_REL);
        while (__atomic_load_n(ctr, __ATOMIC_ACQUIRE) < target)
            __builtin_amdgcn_s_sleep(2);
    }
    __syncthreads();
    __threadfence();
}

// ---------------------------------------------------------------------------
// TDM: load one gate's 16-column x 2048-K bf16 slice of Wt into LDS,
// interleaved as [k_chunk(128)][col(16)][16 elems] via iterate mode:
// per iteration copy a 16(K) x 16(col) tile (tensor_dim0_stride=2048),
// advance global by 16 elems and LDS by one full 64-col chunk row (2048 B).
// clang-23 toolchain: 6-arg builtin (g0 v4u, g1 v8i, g2 v4i, g3 v4i, v8i, cpol)
// ---------------------------------------------------------------------------
#if __has_builtin(__builtin_amdgcn_tensor_load_to_lds)
#define HAVE_TDM 1
__device__ __forceinline__ void tdm_load_gate_slice(const __bf16* gbase,
                                                    unsigned lds_off) {
    unsigned long long ga = (unsigned long long)(uintptr_t)gbase;
    v4u g0;
    g0[0] = 1u;                                          // count=1 (valid)
    g0[1] = lds_off;                                     // lds_addr (bytes)
    g0[2] = (unsigned)(ga & 0xffffffffu);                // global_addr lo
    g0[3] = (unsigned)((ga >> 32) & 0x01ffffffu) | (2u << 30);  // hi | type=2
    v8i g1;
    g1[0] = (1 << 16) | (1 << 19);      // data_size=2B, iterate_enable=1
    g1[1] = (int)(2048u << 16);         // tensor_dim0[15:0]=2048 (K elems)
    g1[2] = (int)(16u << 16);           // tensor_dim1[15:0]=16 rows (cols of W)
    g1[3] = (int)(16u << 16);           // tile_dim0=16 elems (one k-chunk)
    g1[4] = 16;                         // tile_dim1=16 rows
    g1[5] = 2048;                       // tensor_dim0_stride = 2048 elems
    g1[6] = 0;
    g1[7] = 0;
    v4i g2;
    g2[0] = 0;                          // tensor_dim2 (unused, iterate mode)
    g2[1] = 1024;                       // lds_addr_increment = 1024 elems (2KB)
    g2[2] = 16;                         // global_addr_increment = 16 elems
    g2[3] = (int)(127u << 16);          // iterate_count=127 -> 128 iterations
    v4i g3; g3[0] = 0; g3[1] = 0; g3[2] = 0; g3[3] = 0;
    v8i g4; g4[0] = 0; g4[1] = 0; g4[2] = 0; g4[3] = 0;
    g4[4] = 0; g4[5] = 0; g4[6] = 0; g4[7] = 0;
    __builtin_amdgcn_tensor_load_to_lds(g0, g1, g2, g3, g4, 0);
}
#else
#define HAVE_TDM 0
#endif

// ---------------------------------------------------------------------------
// persistent LSTM kernel: 64 WGs x 128 threads (4 wave32)
// wave w = M-tile (batch rows 16w..16w+15); the WG's 16 output units j and
// all 4 gates are handled in-wave (gate combine needs no cross-wave traffic).
// LDS holds the WG's 64 W-columns, chunk-interleaved: [kc][col_local][16].
// ---------------------------------------------------------------------------
__global__ __launch_bounds__(NTHR) void lstm_persistent_kernel(
    const __bf16* __restrict__ Xb,   // [B,S,D] bf16
    const __bf16* __restrict__ Wt,   // [N4][K2] bf16
    const float*  __restrict__ bias, // [N4]
    const float*  __restrict__ lnw,  // [N4]
    const float*  __restrict__ lnb,  // [N4]
    __bf16* __restrict__ Hb,         // [B,D] bf16 hidden state
    float*  __restrict__ Cs,         // [B,D] f32 cell state
    float*  __restrict__ part,       // [NWG][B][2] partial sums
    unsigned* __restrict__ ctr,      // barrier counter
    float*  __restrict__ out)        // [B,S,D]
{
    extern __shared__ __bf16 sW[];   // 64 cols x 2048 K = 256 KB
    __shared__ float sPart[2 * B_];
    __shared__ float sMean[B_];
    __shared__ float sRstd[B_];

    const int wg   = blockIdx.x;
    const int tid  = threadIdx.x;
    const int wave = tid >> 5;
    const int lane = tid & 31;
    const int half = lane >> 4;     // K-group select within fragments
    const int l16  = lane & 15;

    const int m0   = wave * 16;              // batch-row tile base
    const int j0   = wg * JPW;               // output-unit base
    const int jcol = j0 + l16;               // this lane's output unit
    const int rowA = m0 + l16;               // batch row for A loads

    // -------- one-time: stage this WG's W slice into LDS --------
#if HAVE_TDM
    if (tid < 32) {                 // wave 0 issues 4 TDM descriptors
        const unsigned lds_base = (unsigned)(uintptr_t)(void*)sW;
#pragma unroll
        for (int g = 0; g < 4; ++g) {
            const __bf16* gbase = Wt + (size_t)(g * D_ + j0) * K2;
            tdm_load_gate_slice(gbase, lds_base + (unsigned)g * 512u);
        }
        __builtin_amdgcn_s_wait_tensorcnt(0);
    }
#else
    for (int c = tid; c < 64 * 128; c += NTHR) {
        const int col_local = c >> 7;        // 0..63
        const int kc        = c & 127;       // 0..127
        const int gcol = (col_local >> 4) * D_ + j0 + (col_local & 15);
        v16bf v = *(const v16bf*)(Wt + (size_t)gcol * K2 + kc * 16);
        *(v16bf*)(sW + (size_t)(kc * 64 + col_local) * 16) = v;
    }
#endif
    __syncthreads();

    // per-column constants (hoisted out of the time loop)
    float lw[4], lb[4], bs[4];
    const __bf16* sB[4];            // LDS base per gate for this lane
#pragma unroll
    for (int g = 0; g < 4; ++g) {
        const int col = g * D_ + jcol;
        lw[g] = lnw[col];
        lb[g] = lnb[col];
        bs[g] = bias[col];
        // element offset: (k0 + 16*half)*64 + (g*16 + l16)*16
        sB[g] = sW + (size_t)(g * 16 + l16) * 16 + (size_t)half * 1024;
    }

    const __bf16* xrow = Xb + (size_t)rowA * S_ * D_;
    unsigned phase = 0;

    for (int t = 0; t < S_; ++t) {
        if (tid < 2 * B_) sPart[tid] = 0.0f;
        if (t + 1 < S_)
            __builtin_prefetch(xrow + (size_t)(t + 1) * D_, 0, 3);
        __syncthreads();

        v8f acc[4] = {};

        // ---- K loop, x half: A = x_t rows, B from LDS ----
        const __bf16* aX = xrow + (size_t)t * D_;
        for (int k0 = 0; k0 < D_; k0 += 32) {
            AFrag a;
            a.p.lo = *(const v8bf*)(aX + k0 + 16 * half);
            a.p.hi = *(const v8bf*)(aX + k0 + 16 * half + 8);
#pragma unroll
            for (int g = 0; g < 4; ++g) {
                v16bf bm = *(const v16bf*)(sB[g] + (size_t)k0 * 64);
                acc[g] = __builtin_amdgcn_wmma_f32_16x16x32_bf16(
                    false, a.v, false, bm, (short)0, acc[g], false, false);
            }
        }
        // ---- K loop, h half: A = hidden-state rows, B from LDS ----
        const __bf16* aH = Hb + (size_t)rowA * D_;
        for (int k0 = 0; k0 < D_; k0 += 32) {
            AFrag a;
            a.p.lo = *(const v8bf*)(aH + k0 + 16 * half);
            a.p.hi = *(const v8bf*)(aH + k0 + 16 * half + 8);
#pragma unroll
            for (int g = 0; g < 4; ++g) {
                v16bf bm = *(const v16bf*)(sB[g] + (size_t)(k0 + D_) * 64);
                acc[g] = __builtin_amdgcn_wmma_f32_16x16x32_bf16(
                    false, a.v, false, bm, (short)0, acc[g], false, false);
            }
        }

        // ---- add bias; LayerNorm partial sums (per batch row) ----
#pragma unroll
        for (int i = 0; i < 8; ++i) {
            float s = 0.0f, q = 0.0f;
#pragma unroll
            for (int g = 0; g < 4; ++g) {
                float v = acc[g][i] + bs[g];
                acc[g][i] = v;
                s += v;
                q += v * v;
            }
            // reduce across the 16 lanes of each half (rows differ per half)
#pragma unroll
            for (int off = 1; off < 16; off <<= 1) {
                s += __shfl_xor(s, off, 32);
                q += __shfl_xor(q, off, 32);
            }
            if (l16 == 0) {
                const int row = m0 + i + 8 * half;
                atomicAdd(&sPart[row * 2 + 0], s);
                atomicAdd(&sPart[row * 2 + 1], q);
            }
        }
        __syncthreads();
        if (tid < 2 * B_) part[wg * 2 * B_ + tid] = sPart[tid];

        // ---- grid barrier A: all partials visible ----
        ++phase;
        grid_sync(ctr, NWG * phase);

        if (tid < B_) {
            float S = 0.0f, Q = 0.0f;
            for (int w = 0; w < NWG; ++w) {
                S += part[w * 2 * B_ + tid * 2 + 0];
                Q += part[w * 2 * B_ + tid * 2 + 1];
            }
            const float mean = S * (1.0f / N4);
            const float var  = Q * (1.0f / N4) - mean * mean;
            sMean[tid] = mean;
            sRstd[tid] = rsqrtf(var + 1e-5f);
        }
        __syncthreads();

        // ---- normalize, gates, recurrence, publish h ----
#pragma unroll
        for (int i = 0; i < 8; ++i) {
            const int   b    = m0 + i + 8 * half;
            const float mean = sMean[b];
            const float rstd = sRstd[b];
            float gv[4];
#pragma unroll
            for (int g = 0; g < 4; ++g) {
                const float y = (acc[g][i] - mean) * rstd * lw[g] + lb[g];
                gv[g] = (g < 3) ? (1.0f / (1.0f + __expf(-y))) : tanhf(y);
            }
            const size_t sidx = (size_t)b * D_ + jcol;
            float c = Cs[sidx];
            c = gv[1] * c + gv[0] * gv[3];
            const float hnew = gv[2] * tanhf(c);
            Cs[sidx] = c;
            Hb[sidx] = (__bf16)hnew;
            out[((size_t)b * S_ + t) * D_ + jcol] = hnew;
        }

        // ---- grid barrier B: h state published for next step ----
        ++phase;
        grid_sync(ctr, NWG * phase);
    }
}

// ---------------------------------------------------------------------------
// launch
// ---------------------------------------------------------------------------
extern "C" void kernel_launch(void* const* d_in, const int* in_sizes, int n_in,
                              void* d_out, int out_size, void* d_ws, size_t ws_size,
                              hipStream_t stream) {
    const float* X    = (const float*)d_in[0];  // inputo [B,S,D]
    const float* W    = (const float*)d_in[1];  // [2D, 4D]
    const float* bias = (const float*)d_in[2];  // [4D]
    const float* lnw  = (const float*)d_in[3];  // [4D]
    const float* lnb  = (const float*)d_in[4];  // [4D]
    const float* hx   = (const float*)d_in[5];  // [1,D]
    const float* cx   = (const float*)d_in[6];  // [1,D]
    float* out = (float*)d_out;

    char* ws = (char*)d_ws;
    __bf16* Xb = (__bf16*)ws;  ws += (size_t)B_ * S_ * D_ * 2;  // 64 MB
    __bf16* Wt = (__bf16*)ws;  ws += (size_t)N4 * K2 * 2;       // 16 MB
    __bf16* Hb = (__bf16*)ws;  ws += (size_t)B_ * D_ * 2;       // 128 KB
    float*  Cs = (float*)ws;   ws += (size_t)B_ * D_ * 4;       // 256 KB
    float*  part = (float*)ws; ws += (size_t)NWG * 2 * B_ * 4;  // 32 KB
    unsigned* ctr = (unsigned*)ws;

    cvt_x_kernel<<<2048, 256, 0, stream>>>(X, Xb, B_ * S_ * D_);
    cvt_w_kernel<<<2048, 256, 0, stream>>>(W, Wt);
    init_state_kernel<<<256, 256, 0, stream>>>(hx, cx, Hb, Cs, part, ctr);

    const size_t lds_bytes = (size_t)64 * K2 * sizeof(__bf16);  // 256 KB
    lstm_persistent_kernel<<<NWG, NTHR, lds_bytes, stream>>>(
        Xb, Wt, bias, lnw, lnb, Hb, Cs, part, ctr, out);
}
